// GAT_37666863186317
// MI455X (gfx1250) — compile-verified, matrix operand order checked by reading, not run on hardware
//
#include <hip/hip_runtime.h>
#include <math.h>

// ---------- types for WMMA bf16 ----------
typedef __bf16 bf16_t;
typedef bf16_t v16bf __attribute__((ext_vector_type(16)));
typedef float  v8f   __attribute__((ext_vector_type(8)));
typedef unsigned int u32x4 __attribute__((ext_vector_type(4)));

union FragBF { v16bf v; u32x4 u[2]; };

// ---------- problem constants (match reference) ----------
#define C_N0   320000
#define C_N1   80000
#define C_N2   16000
#define C_E1   800000
#define C_E2   160000
#define C_INC  128
#define C_HID  64
#define C_OUTC 48
#define C_H1   4
#define C_HC1  (C_H1 * C_HID)   // 256
#define C_HC2  C_OUTC           // 48
#define NEG_SLOPE 0.2f

// ---------- helpers: order-preserving float<->uint encoding for atomic max ----------
__device__ __forceinline__ unsigned encf(float f) {
    unsigned u = __float_as_uint(f);
    return (u & 0x80000000u) ? ~u : (u | 0x80000000u);
}
__device__ __forceinline__ float decf(unsigned e) {
    return (e & 0x80000000u) ? __uint_as_float(e & 0x7FFFFFFFu) : __uint_as_float(~e);
}

// ---------- conversion kernels ----------
__global__ void cvt_f32_bf16(const float* __restrict__ in, bf16_t* __restrict__ out, long n) {
    long i = ((long)blockIdx.x * blockDim.x + threadIdx.x) * 4;
    if (i + 4 <= n) {
        float4 f = *(const float4*)(in + i);
        out[i + 0] = (bf16_t)f.x;
        out[i + 1] = (bf16_t)f.y;
        out[i + 2] = (bf16_t)f.z;
        out[i + 3] = (bf16_t)f.w;
    } else {
        for (long j = i; j < n; ++j) out[j] = (bf16_t)in[j];
    }
}

// W[K,N] fp32 -> Wt[N,K] bf16 (transposed so B fragments load contiguously)
__global__ void transpose_to_bf16(const float* __restrict__ W, bf16_t* __restrict__ Wt,
                                  int K, int N) {
    int idx = blockIdx.x * blockDim.x + threadIdx.x;
    if (idx >= K * N) return;
    int n = idx / K, k = idx % K;
    Wt[idx] = (bf16_t)W[k * N + n];
}

// ---------- WMMA GEMM: C[M,Ntot] = A[M,K](bf16) @ Bt[Ntot,K]^T(bf16) + bias ----------
// One wave computes a 16 x (NTILES*16) strip. Block = 256 threads = 8 waves.
// A/B fragment layout for v_wmma_f32_16x16x32_bf16 (wave32):
//   lane l: g = l>>4, row/col = l&15; vector elems 0..7  hold K = 8g..8g+7,
//   elems 8..15 hold K = 16+8g..16+8g+7  -> two b128 loads per lane per k-step.
template <int NTILES>
__global__ __launch_bounds__(256) void gemm_bf16_wmma(
    const bf16_t* __restrict__ A,   // [M, K]
    const bf16_t* __restrict__ Bt,  // [Ntot, K]
    const float*  __restrict__ bias,// [Ntot]
    float* __restrict__ C,          // [M, Ntot]
    int M, int K, int Ntot)
{
    const int lane  = threadIdx.x & 31;
    const int wave  = threadIdx.x >> 5;
    const int mtile = blockIdx.x * 8 + wave;
    if (mtile * 16 >= M) return;                 // wave-uniform exit (EXEC stays all-ones)
    const int n0 = blockIdx.y * (NTILES * 16);
    const int g  = lane >> 4;
    const int ml = lane & 15;

    const bf16_t* arow = A + (size_t)(mtile * 16 + ml) * K;

    v8f acc[NTILES] = {};

    for (int kb = 0; kb < K; kb += 32) {
        FragBF a;
        a.u[0] = *(const u32x4*)(arow + kb + 8 * g);
        a.u[1] = *(const u32x4*)(arow + kb + 16 + 8 * g);
#pragma unroll
        for (int t = 0; t < NTILES; ++t) {
            const bf16_t* brow = Bt + (size_t)(n0 + t * 16 + ml) * K;
            FragBF b;
            b.u[0] = *(const u32x4*)(brow + kb + 8 * g);
            b.u[1] = *(const u32x4*)(brow + kb + 16 + 8 * g);
            acc[t] = __builtin_amdgcn_wmma_f32_16x16x32_bf16(
                false, a.v, false, b.v, (short)0, acc[t], false, false);
        }
    }

    // C/D layout: vgpr v, lane l -> M = v + 8*(l>>4), N = l&15
#pragma unroll
    for (int t = 0; t < NTILES; ++t) {
        int n = n0 + t * 16 + ml;
        float bv = bias[n];
#pragma unroll
        for (int v = 0; v < 8; ++v) {
            int m = mtile * 16 + v + 8 * g;
            C[(size_t)m * Ntot + n] = acc[t][v] + bv;
        }
    }
}

// ---------- edge kernels (generic in H, C) ----------
// alpha[e,h] = att[h,:] . leaky_relu(xl[src[e],h,:] + xr[dst[e],h,:]); segment max into amaxEnc
__global__ void edge_logits(const float* __restrict__ xl, const float* __restrict__ xr,
                            const int* __restrict__ src, const int* __restrict__ dst,
                            const float* __restrict__ att,
                            float* __restrict__ alpha, unsigned* __restrict__ amaxEnc,
                            int E, int H, int C)
{
    int e = blockIdx.x * blockDim.x + threadIdx.x;
    if (e >= E) return;
    int s = src[e], d = dst[e];
    const float* ls = xl + (size_t)s * H * C;
    const float* rd = xr + (size_t)d * H * C;
    for (int h = 0; h < H; ++h) {
        const float* lh = ls + h * C;
        const float* rh = rd + h * C;
        const float* ah = att + h * C;
        float a = 0.f;
        for (int c = 0; c < C; c += 4) {
            float4 l4 = *(const float4*)(lh + c);
            float4 r4 = *(const float4*)(rh + c);
            float4 a4 = *(const float4*)(ah + c);
            float e0 = l4.x + r4.x; e0 = e0 > 0.f ? e0 : NEG_SLOPE * e0;
            float e1 = l4.y + r4.y; e1 = e1 > 0.f ? e1 : NEG_SLOPE * e1;
            float e2 = l4.z + r4.z; e2 = e2 > 0.f ? e2 : NEG_SLOPE * e2;
            float e3 = l4.w + r4.w; e3 = e3 > 0.f ? e3 : NEG_SLOPE * e3;
            a += e0 * a4.x + e1 * a4.y + e2 * a4.z + e3 * a4.w;
        }
        alpha[(size_t)e * H + h] = a;
        atomicMax(&amaxEnc[(size_t)d * H + h], encf(a));
    }
}

// alpha <- exp(alpha - amax[dst]); denom[dst] += alpha
__global__ void edge_softmax_num(float* __restrict__ alpha,
                                 const unsigned* __restrict__ amaxEnc,
                                 const int* __restrict__ dst,
                                 float* __restrict__ denom, int E, int H)
{
    int e = blockIdx.x * blockDim.x + threadIdx.x;
    if (e >= E) return;
    int d = dst[e];
    for (int h = 0; h < H; ++h) {
        float m  = decf(amaxEnc[(size_t)d * H + h]);
        float ex = expf(alpha[(size_t)e * H + h] - m);
        alpha[(size_t)e * H + h] = ex;
        atomicAdd(&denom[(size_t)d * H + h], ex);
    }
}

// out[dst,h,:] += xl[src,h,:] * (alpha[e,h] / (denom[dst,h] + 1e-16))
__global__ void edge_aggregate(const float* __restrict__ xl,
                               const float* __restrict__ alpha,
                               const float* __restrict__ denom,
                               const int* __restrict__ src, const int* __restrict__ dst,
                               float* __restrict__ out, int E, int H, int C)
{
    int e = blockIdx.x * blockDim.x + threadIdx.x;
    if (e >= E) return;
    int s = src[e], d = dst[e];
    for (int h = 0; h < H; ++h) {
        float w = alpha[(size_t)e * H + h] / (denom[(size_t)d * H + h] + 1e-16f);
        const float* xs = xl + ((size_t)s * H + h) * C;
        float* od = out + ((size_t)d * H + h) * C;
        for (int c = 0; c < C; c += 4) {
            float4 v = *(const float4*)(xs + c);
            atomicAdd(od + c + 0, v.x * w);
            atomicAdd(od + c + 1, v.y * w);
            atomicAdd(od + c + 2, v.z * w);
            atomicAdd(od + c + 3, v.w * w);
        }
    }
}

// h[i] = relu(h[i] + bias[i % Ncol])  (in place)
__global__ void bias_relu_inplace(float* __restrict__ h, const float* __restrict__ bias,
                                  long n, int Ncol)
{
    long i = (long)blockIdx.x * blockDim.x + threadIdx.x;
    if (i >= n) return;
    float v = h[i] + bias[(int)(i % Ncol)];
    h[i] = v > 0.f ? v : 0.f;
}

// final: out[row,:] = log_softmax(agg[row,:] + bias2)  over 48 cols; one wave per row
__global__ __launch_bounds__(256) void logsoftmax48(const float* __restrict__ agg,
                                                    const float* __restrict__ bias,
                                                    float* __restrict__ out, int rows)
{
    const int lane = threadIdx.x & 31;
    const int wave = threadIdx.x >> 5;
    const int row  = blockIdx.x * 8 + wave;
    if (row >= rows) return;
    const float* r = agg + (size_t)row * C_OUTC;

    float v0 = r[lane] + bias[lane];                        // lane < 32 < 48 always valid
    bool  has1 = (lane + 32) < C_OUTC;
    float v1 = has1 ? (r[lane + 32] + bias[lane + 32]) : -INFINITY;

    float m = fmaxf(v0, v1);
#pragma unroll
    for (int o = 16; o >= 1; o >>= 1) m = fmaxf(m, __shfl_xor(m, o));

    float s = expf(v0 - m) + (has1 ? expf(v1 - m) : 0.f);
#pragma unroll
    for (int o = 16; o >= 1; o >>= 1) s += __shfl_xor(s, o);

    float lg = logf(s);
    out[(size_t)row * C_OUTC + lane] = v0 - m - lg;
    if (has1) out[(size_t)row * C_OUTC + lane + 32] = v1 - m - lg;
}

// ---------- host orchestration ----------
extern "C" void kernel_launch(void* const* d_in, const int* in_sizes, int n_in,
                              void* d_out, int out_size, void* d_ws, size_t ws_size,
                              hipStream_t stream) {
    const float* x     = (const float*)d_in[0];
    const float* Wl1   = (const float*)d_in[1];
    const float* bl1   = (const float*)d_in[2];
    const float* Wr1   = (const float*)d_in[3];
    const float* br1   = (const float*)d_in[4];
    const float* att1  = (const float*)d_in[5];
    const float* bias1 = (const float*)d_in[6];
    const float* Wl2   = (const float*)d_in[7];
    const float* bl2   = (const float*)d_in[8];
    const float* Wr2   = (const float*)d_in[9];
    const float* br2   = (const float*)d_in[10];
    const float* att2  = (const float*)d_in[11];
    const float* bias2 = (const float*)d_in[12];
    const int*   src1  = (const int*)d_in[13];
    const int*   dst1  = (const int*)d_in[14];
    const int*   src2  = (const int*)d_in[15];
    const int*   dst2  = (const int*)d_in[16];
    float* outp = (float*)d_out;

    // workspace carve-out (256B aligned slices)
    char* p = (char*)d_ws;
    auto alloc = [&](size_t bytes) -> void* {
        void* r = (void*)p;
        p += (bytes + 255) & ~(size_t)255;
        return r;
    };
    bf16_t*   xb     = (bf16_t*)  alloc((size_t)C_N0 * C_INC * 2);
    bf16_t*   wl1t   = (bf16_t*)  alloc((size_t)C_HC1 * C_INC * 2);
    bf16_t*   wr1t   = (bf16_t*)  alloc((size_t)C_HC1 * C_INC * 2);
    bf16_t*   wl2t   = (bf16_t*)  alloc((size_t)C_HC2 * C_HC1 * 2);
    bf16_t*   wr2t   = (bf16_t*)  alloc((size_t)C_HC2 * C_HC1 * 2);
    float*    xl1    = (float*)   alloc((size_t)C_N0 * C_HC1 * 4);
    float*    xr1    = (float*)   alloc((size_t)C_N1 * C_HC1 * 4);
    float*    alpha1 = (float*)   alloc((size_t)C_E1 * C_H1 * 4);
    unsigned* amax1  = (unsigned*)alloc((size_t)C_N1 * C_H1 * 4);
    float*    den1   = (float*)   alloc((size_t)C_N1 * C_H1 * 4);
    float*    h1     = (float*)   alloc((size_t)C_N1 * C_HC1 * 4);   // agg -> relu'd h
    bf16_t*   hb     = (bf16_t*)  alloc((size_t)C_N1 * C_HC1 * 2);
    float*    xl2    = (float*)   alloc((size_t)C_N1 * C_HC2 * 4);
    float*    xr2    = (float*)   alloc((size_t)C_N2 * C_HC2 * 4);
    float*    alpha2 = (float*)   alloc((size_t)C_E2 * 4);
    unsigned* amax2  = (unsigned*)alloc((size_t)C_N2 * 4);
    float*    den2   = (float*)   alloc((size_t)C_N2 * 4);
    float*    agg2   = (float*)   alloc((size_t)C_N2 * C_HC2 * 4);

    // ---- precompute: bf16 x, transposed bf16 weights ----
    {
        long n = (long)C_N0 * C_INC;
        cvt_f32_bf16<<<(unsigned)((n / 4 + 255) / 256), 256, 0, stream>>>(x, xb, n);
    }
    transpose_to_bf16<<<(C_INC * C_HC1 + 255) / 256, 256, 0, stream>>>(Wl1, wl1t, C_INC, C_HC1);
    transpose_to_bf16<<<(C_INC * C_HC1 + 255) / 256, 256, 0, stream>>>(Wr1, wr1t, C_INC, C_HC1);
    transpose_to_bf16<<<(C_HC1 * C_HC2 + 255) / 256, 256, 0, stream>>>(Wl2, wl2t, C_HC1, C_HC2);
    transpose_to_bf16<<<(C_HC1 * C_HC2 + 255) / 256, 256, 0, stream>>>(Wr2, wr2t, C_HC1, C_HC2);

    // ---- layer 1 GEMMs (WMMA): xl1 = x@Wl1+bl1 ; xr1 = x[:N1]@Wr1+br1 ----
    {
        dim3 g1(C_N0 / 16 / 8, C_HC1 / (8 * 16));   // (2500, 2)
        gemm_bf16_wmma<8><<<g1, 256, 0, stream>>>(xb, wl1t, bl1, xl1, C_N0, C_INC, C_HC1);
        dim3 g2(C_N1 / 16 / 8, C_HC1 / (8 * 16));   // (625, 2)
        gemm_bf16_wmma<8><<<g2, 256, 0, stream>>>(xb, wr1t, br1, xr1, C_N1, C_INC, C_HC1);
    }

    // ---- layer 1 edge softmax + aggregation ----
    hipMemsetAsync(amax1, 0, (size_t)C_N1 * C_H1 * 4, stream);   // enc(-inf) > 0, so 0 is safe floor
    hipMemsetAsync(den1,  0, (size_t)C_N1 * C_H1 * 4, stream);
    hipMemsetAsync(h1,    0, (size_t)C_N1 * C_HC1 * 4, stream);

    edge_logits<<<(C_E1 + 255) / 256, 256, 0, stream>>>(xl1, xr1, src1, dst1, att1,
                                                        alpha1, amax1, C_E1, C_H1, C_HID);
    edge_softmax_num<<<(C_E1 + 255) / 256, 256, 0, stream>>>(alpha1, amax1, dst1, den1,
                                                             C_E1, C_H1);
    edge_aggregate<<<(C_E1 + 255) / 256, 256, 0, stream>>>(xl1, alpha1, den1, src1, dst1,
                                                           h1, C_E1, C_H1, C_HID);
    {
        long n = (long)C_N1 * C_HC1;
        bias_relu_inplace<<<(unsigned)((n + 255) / 256), 256, 0, stream>>>(h1, bias1, n, C_HC1);
        cvt_f32_bf16<<<(unsigned)((n / 4 + 255) / 256), 256, 0, stream>>>(h1, hb, n);
    }

    // ---- layer 2 GEMMs (WMMA): xl2 = h@Wl2+bl2 ; xr2 = h[:N2]@Wr2+br2 ----
    {
        dim3 g1(C_N1 / 16 / 8, 1);                  // (625, 1), NTILES=3 -> 48 cols
        gemm_bf16_wmma<3><<<g1, 256, 0, stream>>>(hb, wl2t, bl2, xl2, C_N1, C_HC1, C_HC2);
        dim3 g2(C_N2 / 16 / 8, 1);                  // (125, 1)
        gemm_bf16_wmma<3><<<g2, 256, 0, stream>>>(hb, wr2t, br2, xr2, C_N2, C_HC1, C_HC2);
    }

    // ---- layer 2 edge softmax + aggregation ----
    hipMemsetAsync(amax2, 0, (size_t)C_N2 * 4, stream);
    hipMemsetAsync(den2,  0, (size_t)C_N2 * 4, stream);
    hipMemsetAsync(agg2,  0, (size_t)C_N2 * C_HC2 * 4, stream);

    edge_logits<<<(C_E2 + 255) / 256, 256, 0, stream>>>(xl2, xr2, src2, dst2, att2,
                                                        alpha2, amax2, C_E2, 1, C_OUTC);
    edge_softmax_num<<<(C_E2 + 255) / 256, 256, 0, stream>>>(alpha2, amax2, dst2, den2,
                                                             C_E2, 1);
    edge_aggregate<<<(C_E2 + 255) / 256, 256, 0, stream>>>(xl2, alpha2, den2, src2, dst2,
                                                           agg2, C_E2, 1, C_OUTC);

    // ---- final: (+bias2, H=1 mean is identity) then log_softmax -> d_out ----
    logsoftmax48<<<(C_N2 + 7) / 8, 256, 0, stream>>>(agg2, bias2, outp, C_N2);
}
